// Generator_Harm_52578989637887
// MI455X (gfx1250) — matrix operand order repaired and checked by reading, not compiled, not run on hardware
//
#include <hip/hip_runtime.h>
#include <hip/hip_bf16.h>

typedef _Float16 v16h __attribute__((ext_vector_type(16)));
typedef _Float16 v8h  __attribute__((ext_vector_type(8)));
typedef float    v8f  __attribute__((ext_vector_type(8)));

#define HCH   192
#define TLEN  800
#define NHARM 64
#define HOP   256

// ---------------------------------------------------------------------------
// conv1d-as-GEMM, templated so all loop bounds are compile-time.
//   y[b,o,t] = sum_{i,dt} w[o,i,dt] * x[b,i,t+dt-pad] + bias[o]
// Block = (batch b, 16-row time strip), NWAVES waves; wave w owns output
// channels [16w,16w+16).  K ordering: k = i*TAPS + dt  (matches w layout).
//
// Data movement:
//   - activations: global -> LDS via gfx1250 async-to-LDS (raw f32 tile),
//     then expanded once to a f16 A-panel [16][KTOT] in LDS.
//   - weights: each wave preloads ALL its B fragments into VGPRs as f16
//     (KSTEPS x v16h; 144 VGPRs for k3) while the async copy is in flight.
//   - inner loop: 2x ds_load_b128 (A frag) + v_wmma_f32_16x16x32_f16.
// MODE 0: conv+bias   MODE 1: conv+bias+LN+ReLU   MODE 2: conv+bias+resid*mask
// ---------------------------------------------------------------------------
template<int TAPS, int MODE, int NWAVES>
__global__ __launch_bounds__(NWAVES * 32)
void conv_gemm(const float* __restrict__ xin,    // [B][192][T]
               const float* __restrict__ w,      // [NOUT][192][TAPS]
               const float* __restrict__ bias,   // [NOUT]
               const float* __restrict__ gamma,  // [192] (MODE 1)
               const float* __restrict__ beta,   // [192] (MODE 1)
               const float* __restrict__ resid,  // [B][192][T] (MODE 2)
               const float* __restrict__ maskp,  // [B][T]      (MODE 2)
               float* __restrict__ yout,         // [B][NOUT][T]
               int NOUT)
{
    constexpr int KTOT   = HCH * TAPS;       // 576 / 192
    constexpr int KSTEPS = KTOT / 32;        // 18 / 6
    constexpr int AST    = KTOT + 8;         // f16 row stride, 16B aligned
    constexpr int NTHR   = NWAVES * 32;
    constexpr int PAD    = TAPS >> 1;
    constexpr int XCOLS  = 16 + 2 * PAD;     // 18 / 16 staged time columns
    constexpr int XST    = 20;               // f32 row stride (80B)
    static_assert(MODE != 1 || NWAVES == 12, "LN path assumes 384 threads");

    __shared__ float    Xlds[HCH * XST];     // raw f32 activation tile
    __shared__ _Float16 Alds[16 * AST];      // expanded f16 A panel
    __shared__ float    Cs[16 * HCH];
    __shared__ float    Ps[16 * 24];
    __shared__ float    Qs[16 * 24];
    __shared__ float    Mu[16];
    __shared__ float    Rsd[16];

    const int tid  = threadIdx.x;
    const int wave = tid >> 5;
    const int lane = tid & 31;
    const int hi   = lane >> 4;
    const int l16  = lane & 15;
    const int b    = blockIdx.x / 50;
    const int t0   = (blockIdx.x % 50) * 16;

    // ---- 1) async global -> LDS: raw tile x[b, :, t0-PAD .. t0+15+PAD] ----
    for (int idx = tid; idx < HCH * XCOLS; idx += NTHR) {
        int i = idx / XCOLS;
        int j = idx - i * XCOLS;
        int t = t0 - PAD + j;
        if (t >= 0 && t < TLEN) {            // edge lanes masked via EXEC
            int goff = (b * HCH + i) * TLEN + t;                 // element off
            unsigned loff = (unsigned)(unsigned long long)(void*)&Xlds[i * XST + j];
            asm volatile("global_load_async_to_lds_b32 %0, %1, %2 scale_offset"
                         :: "v"(loff), "v"(goff), "s"(xin) : "memory");
        }
    }

    // ---- 2) overlap: preload this wave's full B tile into registers ------
    const int ncol = wave * 16 + l16;
    const int nrow = (ncol < NOUT) ? ncol : 0;   // clamp: no EXEC divergence,
                                                 // junk columns masked at store
    v16h bf[KSTEPS];
    {
        const float4* wrow = (const float4*)(w + (size_t)nrow * KTOT);
        #pragma unroll
        for (int ks = 0; ks < KSTEPS; ++ks) {
            const int base4 = (ks * 32 + hi * 16) >> 2;
            #pragma unroll
            for (int q = 0; q < 4; ++q) {
                float4 f = wrow[base4 + q];
                bf[ks][q * 4 + 0] = (_Float16)f.x;
                bf[ks][q * 4 + 1] = (_Float16)f.y;
                bf[ks][q * 4 + 2] = (_Float16)f.z;
                bf[ks][q * 4 + 3] = (_Float16)f.w;
            }
        }
    }

    // ---- 3) drain async, expand raw tile to f16 A panel (k = i*TAPS+dt) --
    asm volatile("s_wait_asynccnt 0x0" ::: "memory");
    __syncthreads();
    for (int idx = tid; idx < 16 * KTOT; idx += NTHR) {
        int m  = idx / KTOT;
        int k  = idx - m * KTOT;
        int i  = k / TAPS;
        int dt = k - i * TAPS;
        int t  = t0 + m + dt - PAD;
        float v = 0.0f;
        if (t >= 0 && t < TLEN) v = Xlds[i * XST + (m + dt)];
        Alds[m * AST + k] = (_Float16)v;
    }
    __syncthreads();

    // ---- 4) GEMM: fully unrolled K loop, A frags from LDS ----------------
    v8f acc = {0.f, 0.f, 0.f, 0.f, 0.f, 0.f, 0.f, 0.f};
    #pragma unroll
    for (int ks = 0; ks < KSTEPS; ++ks) {
        const int k0 = ks * 32;
        // A frag: lanes 0-15 -> K {k0..k0+15}; lanes 16-31 -> K {k0+16..k0+31}
        v8h a0 = *(const v8h*)&Alds[l16 * AST + k0 + hi * 16];
        v8h a1 = *(const v8h*)&Alds[l16 * AST + k0 + hi * 16 + 8];
        v16h af = __builtin_shufflevector(a0, a1,
                    0,1,2,3,4,5,6,7,8,9,10,11,12,13,14,15);
        acc = __builtin_amdgcn_wmma_f32_16x16x32_f16(false, af, false, bf[ks],
                                                     (short)0, acc, false, false);
    }

    // C/D layout: vgpr r -> M = r + 8*(lane/16), N = lane%16
    if (MODE != 1) {
        if (ncol < NOUT) {
            #pragma unroll
            for (int r = 0; r < 8; ++r) {
                int   mm = r + hi * 8;
                int   t  = t0 + mm;
                float v  = acc[r] + bias[ncol];
                if (MODE == 2) {
                    v += resid[(b * HCH + ncol) * TLEN + t];
                    v *= maskp[b * TLEN + t];
                }
                yout[(b * NOUT + ncol) * TLEN + t] = v;
            }
        }
    } else {
        #pragma unroll
        for (int r = 0; r < 8; ++r) {
            int mm = r + hi * 8;
            Cs[mm * HCH + ncol] = acc[r] + bias[ncol];
        }
        __syncthreads();
        {   // per-row mean/var over 192 channels, 24 partials per row
            int rr = tid / 24;
            int c0 = tid - rr * 24;
            float s = 0.f, s2 = 0.f;
            #pragma unroll
            for (int j = 0; j < 8; ++j) {
                float v = Cs[rr * HCH + c0 + j * 24];
                s += v; s2 += v * v;
            }
            Ps[rr * 24 + c0] = s;
            Qs[rr * 24 + c0] = s2;
        }
        __syncthreads();
        if (tid < 16) {
            float s = 0.f, s2 = 0.f;
            #pragma unroll
            for (int j = 0; j < 24; ++j) { s += Ps[tid * 24 + j]; s2 += Qs[tid * 24 + j]; }
            float mu  = s  * (1.0f / HCH);
            float var = s2 * (1.0f / HCH) - mu * mu;
            Mu[tid]  = mu;
            Rsd[tid] = rsqrtf(var + 1e-5f);
        }
        __syncthreads();
        for (int idx = tid; idx < 16 * HCH; idx += NTHR) {
            int mm = idx & 15;
            int c  = idx >> 4;
            float v = (Cs[mm * HCH + c] - Mu[mm]) * Rsd[mm] * gamma[c] + beta[c];
            v = fmaxf(v, 0.0f);
            yout[(b * HCH + c) * TLEN + t0 + mm] = v;
        }
    }
}

// ---------------------------------------------------------------------------
// Harmonic-amplitude post-processing + per-frame phase step.
// ---------------------------------------------------------------------------
__global__ __launch_bounds__(256)
void param_kernel(const float* __restrict__ post,   // [B][65][T]
                  const float* __restrict__ f0,     // [B][T]
                  float* __restrict__ amps,         // [B][T][64]
                  float* __restrict__ stepo)        // [B][T]
{
    int g = blockIdx.x * blockDim.x + threadIdx.x;
    if (g >= 4 * TLEN) return;
    int b = g / TLEN;
    int t = g - b * TLEN;

    const float LOG10C = 2.302585093f;
    float pitch = f0[b * TLEN + t];

    float v0 = post[(b * 65 + 0) * TLEN + t];
    float s0 = 1.0f / (1.0f + __expf(-v0));
    float total = 2.0f * __expf(LOG10C * __logf(s0)) + 1e-7f;

    float aj[NHARM];
    float sum = 0.f;
    #pragma unroll
    for (int j = 0; j < NHARM; ++j) {
        float v  = post[(b * 65 + 1 + j) * TLEN + t];
        float s  = 1.0f / (1.0f + __expf(-v));
        float p  = 2.0f * __expf(LOG10C * __logf(s)) + 1e-7f;
        float aa = ((pitch * (float)(j + 1)) < 12000.0f ? 1.0f : 0.0f) + 1e-4f;
        aj[j] = p * aa;
        sum  += aj[j];
    }
    float inv = total / sum;
    #pragma unroll
    for (int j = 0; j < NHARM; ++j)
        amps[((size_t)b * TLEN + t) * NHARM + j] = aj[j] * inv;

    stepo[b * TLEN + t] = (6.2831853071795864f / 24000.0f) * pitch;
}

// ---------------------------------------------------------------------------
// Frame-level phase prefix: wave-parallel scan (32 lanes x 25 frames each),
// replaces the serial 800-iteration loop (latency killer).
// prefix[f] = sum_{g<f} HOP*step[g]
// ---------------------------------------------------------------------------
__global__ __launch_bounds__(32)
void prefix_kernel(const float* __restrict__ stepo,
                   float* __restrict__ prefix)
{
    const int b    = blockIdx.x;
    const int lane = threadIdx.x;
    const int f0   = lane * 25;                       // 800 / 32

    float loc[25];
    float lsum = 0.f;
    #pragma unroll
    for (int j = 0; j < 25; ++j) {
        loc[j] = (float)HOP * stepo[b * TLEN + f0 + j];
        lsum  += loc[j];
    }
    // inclusive wave scan of per-lane sums -> exclusive base
    float s = lsum;
    #pragma unroll
    for (int d = 1; d < 32; d <<= 1) {
        float o = __shfl_up(s, d, 32);
        if (lane >= d) s += o;
    }
    float run = s - lsum;                             // exclusive prefix
    #pragma unroll
    for (int j = 0; j < 25; ++j) {
        prefix[b * TLEN + f0 + j] = run;
        run += loc[j];
    }
}

// ---------------------------------------------------------------------------
// Sine bank: one 256-thread block per (b, harmonic, frame) == one HOP.
// 210 MB coalesced stores (~9us HBM floor) + v_sin_f32.
// ---------------------------------------------------------------------------
__global__ __launch_bounds__(256)
void synth_kernel(const float* __restrict__ amps,    // [B][T][64]
                  const float* __restrict__ stepo,   // [B][T]
                  const float* __restrict__ prefix,  // [B][T]
                  float* __restrict__ out)           // [B][64][T*HOP]
{
    int blk = blockIdx.x;
    int f   = blk % TLEN;
    int bh  = blk / TLEN;
    int h   = bh & (NHARM - 1);
    int b   = bh >> 6;

    float st  = stepo[b * TLEN + f];
    float pre = prefix[b * TLEN + f];
    float amp = amps[((size_t)b * TLEN + f) * NHARM + h];

    int   s  = threadIdx.x;
    float ph = (pre + (float)(s + 1) * st) * (float)(h + 1);
    size_t oidx = (size_t)(b * NHARM + h) * (size_t)(TLEN * HOP)
                + (size_t)(f * HOP + s);
    out[oidx] = __sinf(ph) * amp;
}

// ---------------------------------------------------------------------------
extern "C" void kernel_launch(void* const* d_in, const int* in_sizes, int n_in,
                              void* d_out, int out_size, void* d_ws, size_t ws_size,
                              hipStream_t stream)
{
    const float* f0     = (const float*)d_in[0];
    const float* harm   = (const float*)d_in[1];
    const float* maskp  = (const float*)d_in[2];
    const float* pre_w  = (const float*)d_in[3];
    const float* pre_b  = (const float*)d_in[4];
    const float* net_w  = (const float*)d_in[5];
    const float* net_b  = (const float*)d_in[6];
    const float* net_g  = (const float*)d_in[7];
    const float* net_be = (const float*)d_in[8];
    const float* proj_w = (const float*)d_in[9];
    const float* proj_b = (const float*)d_in[10];
    const float* post_w = (const float*)d_in[11];
    const float* post_b = (const float*)d_in[12];
    float* out = (float*)d_out;

    // workspace layout (floats): ~9 MB
    const size_t XSZ = (size_t)4 * HCH * TLEN;     // 614400
    float* ws    = (float*)d_ws;
    float* xbuf  = ws;                             // prenet out / residual
    float* hA    = xbuf + XSZ;
    float* hB    = hA + XSZ;
    float* postb = hB + XSZ;                       // 4*65*800
    float* ampsb = postb + (size_t)4 * 65 * TLEN;
    float* stepb = ampsb + (size_t)4 * TLEN * NHARM;
    float* prefb = stepb + (size_t)4 * TLEN;

    dim3 g200(200);

    // prenet: conv k3
    conv_gemm<3, 0, 12><<<g200, 384, 0, stream>>>(harm, pre_w, pre_b,
        nullptr, nullptr, nullptr, nullptr, xbuf, HCH);
    // 8 residual layers: conv k3 + LN + ReLU
    const float* cur = xbuf;
    for (int l = 0; l < 8; ++l) {
        float* nxt = (l & 1) ? hB : hA;
        conv_gemm<3, 1, 12><<<g200, 384, 0, stream>>>(cur,
            net_w + (size_t)l * HCH * HCH * 3,
            net_b + l * HCH, net_g + l * HCH, net_be + l * HCH,
            nullptr, nullptr, nxt, HCH);
        cur = nxt;
    }
    // proj (k1) + residual + mask
    conv_gemm<1, 2, 12><<<g200, 384, 0, stream>>>(cur, proj_w, proj_b,
        nullptr, nullptr, xbuf, maskp, hA, HCH);
    // post conv k3 -> 65 channels (5 waves)
    conv_gemm<3, 0, 5><<<g200, 160, 0, stream>>>(hA, post_w, post_b,
        nullptr, nullptr, nullptr, nullptr, postb, 65);
    // amplitude post-processing + phase steps
    param_kernel<<<(4 * TLEN + 255) / 256, 256, 0, stream>>>(postb, f0, ampsb, stepb);
    prefix_kernel<<<4, 32, 0, stream>>>(stepb, prefb);
    // sine bank
    synth_kernel<<<4 * NHARM * TLEN, 256, 0, stream>>>(ampsb, stepb, prefb, out);
}
